// Attention_38113539785237
// MI455X (gfx1250) — compile-verified, matrix operand order checked by reading, not run on hardware
//
#include <hip/hip_runtime.h>
#include <hip/hip_bf16.h>
#include <math.h>

typedef __attribute__((ext_vector_type(16))) __bf16 v16bf;
typedef __attribute__((ext_vector_type(8)))  float  v8f;

#define H 512
#define B 32
#define S 4096

static __device__ __forceinline__ unsigned pack_bf16(float lo, float hi) {
    union { __bf16 h[2]; unsigned u; } p;
    p.h[0] = (__bf16)lo;
    p.h[1] = (__bf16)hi;
    return p.u;
}

// ---------------------------------------------------------------------------
// Kernel 1: q = query @ Wq^T + bq  (32x512, trivial)  + zero-init of c
// ---------------------------------------------------------------------------
__global__ __launch_bounds__(256) void qproj_kernel(
    const float* __restrict__ query, const float* __restrict__ Wq,
    const float* __restrict__ bq, float* __restrict__ q, float* __restrict__ c)
{
    int idx = blockIdx.x * 256 + threadIdx.x;   // 0..16383  (b*512 + o)
    int b = idx >> 9, o = idx & 511;
    float acc = bq[o];
    const float* qr = &query[b * H];
    const float* wr = &Wq[(size_t)o * H];
    for (int k = 0; k < H; ++k) acc += qr[k] * wr[k];
    q[idx] = acc;
    c[idx] = 0.0f;
}

// ---------------------------------------------------------------------------
// Kernel 2: fused  r = ref@Wr^T + br  (bf16 WMMA, f32 accum)  and
//           logits[b,s] = sum_o V[o]*tanh(q[b,o] + r[s,o])
// One WG per (64 s-rows, batch b). 8 wave32s: 4 s-blocks x 2 o-halves.
// Staging: float2 global loads -> packed bf16 pair -> ds_store_b32.
// ---------------------------------------------------------------------------
__global__ __launch_bounds__(256) void attn_score_kernel(
    const float* __restrict__ ref, const float* __restrict__ Wr,
    const float* __restrict__ br,  const float* __restrict__ V,
    const float* __restrict__ q,   float* __restrict__ logits)
{
    __shared__ unsigned As_u[64 * 16];    // 4 KB  : ref tile  (s_local x 32k as bf16 pairs)
    __shared__ unsigned Bs_u[H * 16];     // 32 KB : Wr panel  (o x 32k as bf16 pairs)
    __shared__ float    lds_logits[64];

    const __bf16* As = (const __bf16*)As_u;
    const __bf16* Bs = (const __bf16*)Bs_u;

    const int b    = blockIdx.y;
    const int s0   = blockIdx.x * 64;
    const int tid  = threadIdx.x;
    const int wave = tid >> 5;
    const int lane = tid & 31;
    const int sb   = wave & 3;            // which 16 s-rows
    const int oh   = wave >> 2;           // 0 or 1: which 256 o-cols
    const int n    = lane & 15;           // N / M index within tile
    const int kh   = (lane >> 4) * 8;     // K half select per ISA bf16 layout

    if (tid < 64) lds_logits[tid] = 0.0f;

    v8f acc[16];
#pragma unroll
    for (int j = 0; j < 16; ++j) acc[j] = (v8f){0.f,0.f,0.f,0.f,0.f,0.f,0.f,0.f};

    const size_t ref_base = ((size_t)b * S + s0) * H;

    for (int k0 = 0; k0 < H; k0 += 32) {
        __syncthreads();
        // stage A: 64x32 fp32 -> packed bf16 (1024 uints, 4 per thread)
#pragma unroll
        for (int t = 0; t < 4; ++t) {
            int idx = tid + t * 256;               // 0..1023
            int row = idx >> 4, kp = idx & 15;     // kp = pair index
            const float2 f = *(const float2*)&ref[ref_base + (size_t)row * H + k0 + kp * 2];
            As_u[idx] = pack_bf16(f.x, f.y);
        }
        // stage B: 512x32 fp32 -> packed bf16 (8192 uints, 32 per thread)
        for (int t = 0; t < 32; ++t) {
            int idx = tid + t * 256;               // 0..8191
            int o = idx >> 4, kp = idx & 15;
            const float2 f = *(const float2*)&Wr[(size_t)o * H + k0 + kp * 2];
            Bs_u[idx] = pack_bf16(f.x, f.y);
        }
        __syncthreads();

        // A fragment: 16x32 bf16, lane n = row M, K split per 16-bit A layout
        v16bf afrag;
        {
            const __bf16* ap = &As[(sb * 16 + n) * 32];
#pragma unroll
            for (int e = 0; e < 8; ++e) afrag[e]     = ap[kh + e];
#pragma unroll
            for (int e = 0; e < 8; ++e) afrag[8 + e] = ap[16 + kh + e];
        }

#pragma unroll
        for (int j = 0; j < 16; ++j) {
            const __bf16* bp = &Bs[(oh * 256 + j * 16 + n) * 32];
            v16bf bfrag;
#pragma unroll
            for (int e = 0; e < 8; ++e) bfrag[e]     = bp[kh + e];
#pragma unroll
            for (int e = 0; e < 8; ++e) bfrag[8 + e] = bp[16 + kh + e];
            acc[j] = __builtin_amdgcn_wmma_f32_16x16x32_bf16(
                false, afrag, false, bfrag, (short)0, acc[j], false, false);
        }
    }

    // Epilogue: C/D layout => VGPR i holds rows M=i (lanes 0-15) / M=i+8
    // (lanes 16-31), column N = lane%16. Reduce V[o]*tanh(.) over o.
    float p[8];
#pragma unroll
    for (int i = 0; i < 8; ++i) p[i] = 0.0f;
#pragma unroll
    for (int j = 0; j < 16; ++j) {
        int o = oh * 256 + j * 16 + n;
        float base = q[b * H + o] + br[o];
        float Vv = V[o];
#pragma unroll
        for (int i = 0; i < 8; ++i)
            p[i] += Vv * tanhf(base + acc[j][i]);
    }
#pragma unroll
    for (int i = 0; i < 8; ++i) {
        float v = p[i];
        v += __shfl_xor(v, 1, 32);
        v += __shfl_xor(v, 2, 32);
        v += __shfl_xor(v, 4, 32);
        v += __shfl_xor(v, 8, 32);
        if (n == 0)  // lane 0 -> rows 0..7, lane 16 -> rows 8..15
            atomicAdd(&lds_logits[sb * 16 + (lane >> 4) * 8 + i], v);
    }
    __syncthreads();
    if (tid < 64) logits[(size_t)b * S + s0 + tid] = lds_logits[tid];
}

// ---------------------------------------------------------------------------
// Kernel 3: in-place softmax over S per batch (one WG per b)
// ---------------------------------------------------------------------------
__global__ __launch_bounds__(256) void softmax_kernel(float* __restrict__ logits)
{
    __shared__ float red[256];
    const int b = blockIdx.x, tid = threadIdx.x;
    float* row = logits + (size_t)b * S;
    float vals[16];
    float m = -INFINITY;
#pragma unroll
    for (int t = 0; t < 16; ++t) { vals[t] = row[tid + t * 256]; m = fmaxf(m, vals[t]); }
    red[tid] = m; __syncthreads();
    for (int off = 128; off > 0; off >>= 1) {
        if (tid < off) red[tid] = fmaxf(red[tid], red[tid + off]);
        __syncthreads();
    }
    m = red[0]; __syncthreads();
    float s = 0.0f;
#pragma unroll
    for (int t = 0; t < 16; ++t) { vals[t] = __expf(vals[t] - m); s += vals[t]; }
    red[tid] = s; __syncthreads();
    for (int off = 128; off > 0; off >>= 1) {
        if (tid < off) red[tid] += red[tid + off];
        __syncthreads();
    }
    float inv = 1.0f / red[0];
#pragma unroll
    for (int t = 0; t < 16; ++t) row[tid + t * 256] = vals[t] * inv;
}

// ---------------------------------------------------------------------------
// Kernel 4: c[b,:] = sum_s w[b,s] * ref[b,s,:]   (memory-bound, 256 MB read)
// grid (B, 8): each WG handles 512 s-rows, accumulates via f32 atomics.
// ---------------------------------------------------------------------------
__global__ __launch_bounds__(256) void wsum_kernel(
    const float* __restrict__ ref, const float* __restrict__ w,
    float* __restrict__ c)
{
    const int b = blockIdx.x, sy = blockIdx.y, tid = threadIdx.x;
    float a0 = 0.0f, a1 = 0.0f;
    const float* wrow  = w + (size_t)b * S + sy * 512;
    const float* rbase = ref + ((size_t)b * S + (size_t)sy * 512) * H;
    for (int s = 0; s < 512; ++s) {
        float ww = wrow[s];
        const float* r = rbase + (size_t)s * H;
        a0 += ww * r[tid];
        a1 += ww * r[tid + 256];
    }
    atomicAdd(&c[b * H + tid], a0);
    atomicAdd(&c[b * H + tid + 256], a1);
}

// ---------------------------------------------------------------------------
// Kernel 5: out[b,h] = c[b,:] . Wr[h,:] + br[h]   (weights sum to 1 => exact)
// ---------------------------------------------------------------------------
__global__ __launch_bounds__(256) void outproj_kernel(
    const float* __restrict__ c, const float* __restrict__ Wr,
    const float* __restrict__ br, float* __restrict__ out)
{
    int idx = blockIdx.x * 256 + threadIdx.x;   // b*512 + h
    int b = idx >> 9, h = idx & 511;
    float acc = br[h];
    const float* cr = &c[b * H];
    const float* wr = &Wr[(size_t)h * H];
    for (int k = 0; k < H; ++k) acc += cr[k] * wr[k];
    out[idx] = acc;                             // out shape [B,H,1] flat
}

extern "C" void kernel_launch(void* const* d_in, const int* in_sizes, int n_in,
                              void* d_out, int out_size, void* d_ws, size_t ws_size,
                              hipStream_t stream) {
    const float* query = (const float*)d_in[0];
    const float* ref   = (const float*)d_in[1];
    const float* Wq    = (const float*)d_in[2];
    const float* bq    = (const float*)d_in[3];
    const float* Wr    = (const float*)d_in[4];
    const float* br    = (const float*)d_in[5];
    const float* V     = (const float*)d_in[6];
    float* out = (float*)d_out;

    float* ws     = (float*)d_ws;
    float* q      = ws;                 // B*H
    float* c      = ws + B * H;         // B*H
    float* logits = ws + 2 * B * H;     // B*S (softmax done in place)

    qproj_kernel     <<<64, 256, 0, stream>>>(query, Wq, bq, q, c);
    attn_score_kernel<<<dim3(S / 64, B), 256, 0, stream>>>(ref, Wr, br, V, q, logits);
    softmax_kernel   <<<B, 256, 0, stream>>>(logits);
    wsum_kernel      <<<dim3(B, 8), 256, 0, stream>>>(ref, logits, c);
    outproj_kernel   <<<64, 256, 0, stream>>>(c, Wr, br, out);
}
